// GraphAttentionNetwork_12584254177943
// MI455X (gfx1250) — matrix operand order repaired and checked by reading, not compiled, not run on hardware
//
#include <hip/hip_runtime.h>
#include <hip/hip_bf16.h>

typedef __attribute__((ext_vector_type(16))) _Float16 v16h;
typedef __attribute__((ext_vector_type(8)))  _Float16 v8h;
typedef __attribute__((ext_vector_type(8)))  float    v8f;

#define D_IN   256
#define NHEADS 8
#define DPH    32
#define MTOT   1312   // 512 Q + 512 K + 32 V + 256 base (X@Wb.T)
#define MPAD   1344   // padded to 21 tiles of 64 columns
#define QOFF   0
#define KOFF   512
#define VOFF   1024
#define BOFF   1056

// ---------- helpers: sortable-uint encoding for float atomicMax ----------
__device__ __forceinline__ unsigned f2sort(float f) {
    unsigned u = __float_as_uint(f);
    return (u & 0x80000000u) ? ~u : (u | 0x80000000u);
}
__device__ __forceinline__ float sort2f(unsigned u) {
    return (u & 0x80000000u) ? __uint_as_float(u & 0x7FFFFFFFu)
                             : __uint_as_float(~u);
}

// ---------- phase 0: convert features f32 -> f16 ----------
__global__ void k_cvt_feats(const float* __restrict__ x, _Float16* __restrict__ xh,
                            long long n) {
    long long t = (long long)blockIdx.x * blockDim.x + threadIdx.x;
    if (t < n) xh[t] = (_Float16)x[t];
}

// ---------- phase 0b: pack all weights into Wall[MPAD][256] f16 + ball[MPAD] ----------
__global__ void k_pack_w(const float* __restrict__ Wq, const float* __restrict__ bq,
                         const float* __restrict__ Wk, const float* __restrict__ bk,
                         const float* __restrict__ Ww, const float* __restrict__ bw,
                         const float* __restrict__ Wb, const float* __restrict__ bb,
                         _Float16* __restrict__ Wall, float* __restrict__ ball) {
    int t = blockIdx.x * blockDim.x + threadIdx.x;
    if (t >= MPAD * D_IN) return;
    int r = t / D_IN, k = t % D_IN;
    float wv = 0.f, bv = 0.f;
    if (r < 512)       { wv = Wq[(long long)r * D_IN + k];          bv = bq[r];        }
    else if (r < 1024) { wv = Wk[(long long)(r - 512) * D_IN + k];  bv = bk[r - 512];  }
    else if (r < 1056) { wv = Ww[(long long)(r - 1024) * D_IN + k]; bv = bw[r - 1024]; }
    else if (r < 1312) { wv = Wb[(long long)(r - 1056) * D_IN + k]; bv = bb[r - 1056]; }
    Wall[t] = (_Float16)wv;
    if (k == 0) ball[r] = bv;
}

// ---------- phase 0c: zero softmax stats ----------
__global__ void k_init_stats(unsigned* __restrict__ umax, float* __restrict__ denom, int n) {
    int t = blockIdx.x * blockDim.x + threadIdx.x;
    if (t < n) { umax[t] = 0u; denom[t] = 0.f; }  // 0 < f2sort(-inf)=0x007FFFFF: identity for max
}

// ---------- phase 1: fused WMMA GEMM  C[N,1312] = Xh @ Wall.T + ball ----------
// One wave computes a 16(M) x 64(N) tile: 4 accumulators share each A fragment.
__global__ void k_gemm_wmma(const _Float16* __restrict__ Xh,
                            const _Float16* __restrict__ Wall,
                            const float* __restrict__ ball,
                            float* __restrict__ Cbuf,
                            int n_nodes, int m_tiles) {
    const int NT64 = MPAD / 64;  // 21
    const int wave = blockIdx.x * (blockDim.x >> 5) + (threadIdx.x >> 5);
    const int lane = threadIdx.x & 31;
    const int m_tile = wave / NT64;
    const int n_tile = wave % NT64;
    if (m_tile >= m_tiles) return;

    const int hi = lane >> 4;          // lane half selects K sub-block (k = 16*hi + s, same order in A and B)
    int row = m_tile * 16 + (lane & 15);
    if (row >= n_nodes) row = n_nodes - 1;           // clamp (N divisible by 16 anyway)
    const _Float16* arow = Xh + (long long)row * D_IN + (hi ? 16 : 0);

    const int ccol = n_tile * 64 + (lane & 15);      // output channel handled by lane, sub-tile 0
    const _Float16* b0 = Wall + (long long)(ccol +  0) * D_IN + (hi ? 16 : 0);
    const _Float16* b1 = Wall + (long long)(ccol + 16) * D_IN + (hi ? 16 : 0);
    const _Float16* b2 = Wall + (long long)(ccol + 32) * D_IN + (hi ? 16 : 0);
    const _Float16* b3 = Wall + (long long)(ccol + 48) * D_IN + (hi ? 16 : 0);

    v8f acc0 = {}, acc1 = {}, acc2 = {}, acc3 = {};

    #pragma unroll
    for (int k0 = 0; k0 < D_IN; k0 += 32) {
        // A fragment: 16 contiguous halves K k0 + 16*hi + {0..15}
        v8h alo = *(const v8h*)(arow + k0);
        v8h ahi = *(const v8h*)(arow + k0 + 8);
        v16h a;
        #pragma unroll
        for (int i = 0; i < 8; ++i) { a[i] = alo[i]; a[i + 8] = ahi[i]; }

        // B fragments: 16 contiguous halves of W[out_ch, k0 + 16*hi ..]  (computes X @ W.T)
        v16h vb0 = *(const v16h*)(b0 + k0);
        v16h vb1 = *(const v16h*)(b1 + k0);
        v16h vb2 = *(const v16h*)(b2 + k0);
        v16h vb3 = *(const v16h*)(b3 + k0);

        acc0 = __builtin_amdgcn_wmma_f32_16x16x32_f16(false, a, false, vb0, (short)0, acc0, false, false);
        acc1 = __builtin_amdgcn_wmma_f32_16x16x32_f16(false, a, false, vb1, (short)0, acc1, false, false);
        acc2 = __builtin_amdgcn_wmma_f32_16x16x32_f16(false, a, false, vb2, (short)0, acc2, false, false);
        acc3 = __builtin_amdgcn_wmma_f32_16x16x32_f16(false, a, false, vb3, (short)0, acc3, false, false);
    }

    // D layout: lanes 0-15 -> M = j, lanes 16-31 -> M = j+8; N = lane&15
    #pragma unroll
    for (int j = 0; j < 8; ++j) {
        int m = m_tile * 16 + j + (hi ? 8 : 0);
        if (m >= n_nodes) continue;
        float* crow = Cbuf + (long long)m * MTOT;
        int c = ccol;
        if (c      < MTOT) crow[c     ] = acc0[j] + ball[c     ];
        if (c + 16 < MTOT) crow[c + 16] = acc1[j] + ball[c + 16];
        if (c + 32 < MTOT) crow[c + 32] = acc2[j] + ball[c + 32];
        if (c + 48 < MTOT) crow[c + 48] = acc3[j] + ball[c + 48];
    }
}

// ---------- phase 2: per-(edge,head) score + segment max ----------
__global__ void k_edge_scores(const float* __restrict__ Cbuf, const int* __restrict__ edges,
                              long long E, unsigned* __restrict__ umax,
                              float* __restrict__ sbuf) {
    long long t = (long long)blockIdx.x * blockDim.x + threadIdx.x;
    if (t >= E * NHEADS) return;
    long long e = t >> 3;
    int h = (int)(t & 7);
    int src = edges[e], dst = edges[E + e];
    const float* q = Cbuf + (long long)src * MTOT + QOFF + h * 64;
    const float* k = Cbuf + (long long)dst * MTOT + KOFF + h * 64;
    float s = 0.f;
    #pragma unroll 8
    for (int d = 0; d < 64; ++d) s += q[d] * k[d];
    sbuf[t] = s;
    atomicMax(&umax[(long long)src * NHEADS + h], f2sort(s));
}

// ---------- phase 3: exp(s - max) + segment sum ----------
__global__ void k_edge_exp(const int* __restrict__ edges, long long E,
                           const unsigned* __restrict__ umax,
                           float* __restrict__ denom, float* __restrict__ sbuf) {
    long long t = (long long)blockIdx.x * blockDim.x + threadIdx.x;
    if (t >= E * NHEADS) return;
    long long e = t >> 3;
    int h = (int)(t & 7);
    int src = edges[e];
    float m = sort2f(umax[(long long)src * NHEADS + h]);
    float ex = __expf(sbuf[t] - m);
    sbuf[t] = ex;
    atomicAdd(&denom[(long long)src * NHEADS + h], ex);
}

// ---------- phase 4: d_out = base projection (X@Wb.T + bb) ----------
__global__ void k_base_out(const float* __restrict__ Cbuf, float* __restrict__ out,
                           long long n) {
    long long t = (long long)blockIdx.x * blockDim.x + threadIdx.x;
    if (t >= n) return;
    long long node = t >> 8;             // /256
    int c = (int)(t & 255);
    out[t] = Cbuf[node * MTOT + BOFF + c];
}

// ---------- phase 5: aggregation — 32 lanes (one per V channel) per (edge,head) ----------
__global__ void k_agg(const float* __restrict__ Cbuf, const int* __restrict__ edges,
                      long long E, const float* __restrict__ denom,
                      const float* __restrict__ sbuf, float* __restrict__ out) {
    long long t = (long long)blockIdx.x * blockDim.x + threadIdx.x;
    if (t >= E * NHEADS * DPH) return;
    long long pair = t >> 5;             // (edge,head)
    int c = (int)(t & 31);
    long long e = pair >> 3;
    int h = (int)(pair & 7);
    int src = edges[e], dst = edges[E + e];
    float alpha = sbuf[pair] / denom[(long long)src * NHEADS + h];
    float v = Cbuf[(long long)dst * MTOT + VOFF + c];
    atomicAdd(&out[(long long)src * 256 + h * DPH + c], alpha * v);
}

extern "C" void kernel_launch(void* const* d_in, const int* in_sizes, int n_in,
                              void* d_out, int out_size, void* d_ws, size_t ws_size,
                              hipStream_t stream) {
    const float* features = (const float*)d_in[0];
    const int*   edges    = (const int*)  d_in[1];
    const float* Wq = (const float*)d_in[2];
    const float* bq = (const float*)d_in[3];
    const float* Wk = (const float*)d_in[4];
    const float* bk = (const float*)d_in[5];
    const float* Ww = (const float*)d_in[6];
    const float* bw = (const float*)d_in[7];
    const float* Wb = (const float*)d_in[8];
    const float* bb = (const float*)d_in[9];
    float* out = (float*)d_out;

    const long long N = in_sizes[0] / D_IN;       // 50000 nodes
    const long long E = in_sizes[1] / 2;          // edge count

    // ---- carve workspace (256B aligned) ----
    char* p = (char*)d_ws;
    auto carve = [&](size_t bytes) -> void* {
        void* r = (void*)p;
        p += (bytes + 255) & ~(size_t)255;
        return r;
    };
    _Float16* Xh   = (_Float16*)carve((size_t)N * D_IN * sizeof(_Float16));
    _Float16* Wall = (_Float16*)carve((size_t)MPAD * D_IN * sizeof(_Float16));
    float*    ball = (float*)   carve((size_t)MPAD * sizeof(float));
    float*    Cbuf = (float*)   carve((size_t)N * MTOT * sizeof(float));
    unsigned* umax = (unsigned*)carve((size_t)N * NHEADS * sizeof(unsigned));
    float*    denom= (float*)   carve((size_t)N * NHEADS * sizeof(float));
    float*    sbuf = (float*)   carve((size_t)E * NHEADS * sizeof(float));
    (void)ws_size;

    const int B = 256;

    // 0) conversions + init
    long long nfeat = N * D_IN;
    k_cvt_feats<<<(unsigned)((nfeat + B - 1) / B), B, 0, stream>>>(features, Xh, nfeat);
    k_pack_w<<<(MPAD * D_IN + B - 1) / B, B, 0, stream>>>(Wq, bq, Wk, bk, Ww, bw, Wb, bb, Wall, ball);
    int nstat = (int)(N * NHEADS);
    k_init_stats<<<(nstat + B - 1) / B, B, 0, stream>>>(umax, denom, nstat);

    // 1) fused WMMA GEMM over all projections
    int m_tiles = (int)((N + 15) / 16);
    long long waves = (long long)m_tiles * (MPAD / 64);
    unsigned gemm_blocks = (unsigned)((waves + 7) / 8);   // 8 waves / 256-thread block
    k_gemm_wmma<<<gemm_blocks, B, 0, stream>>>(Xh, Wall, ball, Cbuf, (int)N, m_tiles);

    // 2-3) edge scores + segment softmax
    long long eh = E * NHEADS;
    k_edge_scores<<<(unsigned)((eh + B - 1) / B), B, 0, stream>>>(Cbuf, edges, E, umax, sbuf);
    k_edge_exp<<<(unsigned)((eh + B - 1) / B), B, 0, stream>>>(edges, E, umax, denom, sbuf);

    // 4) base output
    long long nout = N * 256;
    k_base_out<<<(unsigned)((nout + B - 1) / B), B, 0, stream>>>(Cbuf, out, nout);

    // 5) alpha-weighted aggregation into d_out
    long long aggt = eh * DPH;
    k_agg<<<(unsigned)((aggt + B - 1) / B), B, 0, stream>>>(Cbuf, edges, E, denom, sbuf, out);
}
